// SWAChunk1_79164837199873
// MI455X (gfx1250) — matrix operand order repaired and checked by reading, not compiled, not run on hardware
//
#include <hip/hip_runtime.h>

// ---------------- model constants ----------------
#define H     2048
#define NH    8
#define NKV   2
#define HD    256
#define NREP  4
#define DFF   4096
#define PLD   256
#define NLT   35
#define NL    8
#define WIN   512
#define FULL  4096
#define EPS   1e-6f
#define SPLIT 64          // keys per attention split block

typedef __attribute__((ext_vector_type(16))) _Float16 v16h;
typedef __attribute__((ext_vector_type(8)))  _Float16 v8h;
typedef __attribute__((ext_vector_type(8)))  float    v8f;

// ---------------- helpers ----------------
__device__ __forceinline__ float gelu_tanh(float x) {
    float x3 = x * x * x;
    return 0.5f * x * (1.0f + tanhf(0.7978845608028654f * (x + 0.044715f * x3)));
}

// block-wide sum for blockDim.x == 256 (8 waves of 32)
__device__ __forceinline__ float block_sum_256(float v, float* red) {
    for (int off = 16; off > 0; off >>= 1) v += __shfl_down(v, off, 32);
    int tid = threadIdx.x;
    if ((tid & 31) == 0) red[tid >> 5] = v;
    __syncthreads();
    if (tid == 0) {
        float t = 0.f;
        for (int i = 0; i < 8; ++i) t += red[i];
        red[0] = t;
    }
    __syncthreads();
    return red[0];
}

// ---------------- elementwise / small kernels ----------------
__global__ __launch_bounds__(256) void copy_kernel(float* __restrict__ dst,
                                                   const float* __restrict__ src, int n) {
    int i = blockIdx.x * 256 + threadIdx.x;
    if (i < n) dst[i] = src[i];
}

__global__ __launch_bounds__(256) void gelu_mul_kernel(float* __restrict__ out,
                                                       const float* __restrict__ a,
                                                       const float* __restrict__ b, int n) {
    int i = blockIdx.x * 256 + threadIdx.x;
    if (i < n) out[i] = gelu_tanh(a[i]) * b[i];
}

// out = in * rsqrt(mean(in^2)+eps) * w        (single block, n multiple of 256)
__global__ __launch_bounds__(256) void rms_kernel(float* __restrict__ out,
                                                  const float* __restrict__ in,
                                                  const float* __restrict__ w, int n) {
    __shared__ float red[8];
    float ss = 0.f;
    for (int i = threadIdx.x; i < n; i += 256) { float v = in[i]; ss += v * v; }
    float tot = block_sum_256(ss, red);
    float r = rsqrtf(tot / (float)n + EPS);
    for (int i = threadIdx.x; i < n; i += 256) out[i] = in[i] * r * w[i];
}

// hs = (hs + rms(in,w)) * sc   (single block; sc read from device or 1)
__global__ __launch_bounds__(256) void addrms_kernel(float* __restrict__ hs,
                                                     const float* __restrict__ in,
                                                     const float* __restrict__ w, int n,
                                                     const float* __restrict__ scp, int sci) {
    __shared__ float red[8];
    float ss = 0.f;
    for (int i = threadIdx.x; i < n; i += 256) { float v = in[i]; ss += v * v; }
    float tot = block_sum_256(ss, red);
    float r  = rsqrtf(tot / (float)n + EPS);
    float sc = scp ? scp[sci] : 1.0f;
    for (int i = threadIdx.x; i < n; i += 256) hs[i] = (hs[i] + in[i] * r * w[i]) * sc;
}

// per-layer-embedding: plc = (rms(proj)*plw + raw) * 2^-0.5   (35 blocks x 256)
__global__ __launch_bounds__(256) void ple_kernel(float* __restrict__ plc,
                                                  const float* __restrict__ proj,
                                                  const float* __restrict__ plw,
                                                  const float* __restrict__ raw) {
    __shared__ float red[8];
    int t = blockIdx.x, d = threadIdx.x;
    float x = proj[t * PLD + d];
    float tot = block_sum_256(x * x, red);
    float r = rsqrtf(tot / (float)PLD + EPS);
    plc[t * PLD + d] = (x * r * plw[d] + raw[t * PLD + d]) * 0.7071067811865476f;
}

// q/k rms-norm + rope, v norm. blocks: 0..7 q heads, 8..9 k heads, 10..11 v heads
__global__ __launch_bounds__(256) void qkvpost_kernel(float* __restrict__ q,
                                                      float* __restrict__ k,
                                                      float* __restrict__ v,
                                                      const float* __restrict__ qw,
                                                      const float* __restrict__ kw,
                                                      const float* __restrict__ cosv,
                                                      const float* __restrict__ sinv) {
    __shared__ float red[8];
    __shared__ float buf[HD];
    int b = blockIdx.x, d = threadIdx.x;
    float* vec; const float* w; int dorope;
    if (b < 8)       { vec = q + b * HD;        w = qw;      dorope = 1; }
    else if (b < 10) { vec = k + (b - 8) * HD;  w = kw;      dorope = 1; }
    else             { vec = v + (b - 10) * HD; w = nullptr; dorope = 0; }
    float x = vec[d];
    float tot = block_sum_256(x * x, red);
    float r  = rsqrtf(tot / (float)HD + EPS);
    float nv = x * r * (w ? w[d] : 1.0f);
    if (dorope) {
        buf[d] = nv;
        __syncthreads();
        float rot = (d < HD / 2) ? -buf[d + HD / 2] : buf[d - HD / 2];
        nv = nv * cosv[d] + rot * sinv[d];
    }
    vec[d] = nv;
}

// ---------------- GEMV: one row per wave32, float4 streaming ----------------
__global__ __launch_bounds__(256) void gemv_kernel(float* __restrict__ y,
                                                   const float* __restrict__ x,
                                                   const float* __restrict__ W,
                                                   int K, float scale) {
    const int wave = threadIdx.x >> 5, lane = threadIdx.x & 31;
    const int row  = blockIdx.x * 8 + wave;
    const float4* Wr = (const float4*)(W + (size_t)row * K);
    const float4* x4 = (const float4*)x;
    const int n4 = K >> 2;
    float s = 0.f;
    for (int j = lane; j < n4; j += 32) {
        __builtin_prefetch((const void*)(Wr + j + 64), 0, 1);  // global_prefetch_b8
        float4 w = Wr[j], xv = x4[j];
        s += w.x * xv.x + w.y * xv.y + w.z * xv.z + w.w * xv.w;
    }
    for (int off = 16; off > 0; off >>= 1) s += __shfl_down(s, off, 32);
    if (lane == 0) y[row] = s * scale;
}

// ---------------- KV cache update (writes final outputs) ----------------
// sliding: dst[h][p] = src[h][p+1] for p<WIN-1, new row at p==WIN-1
__global__ __launch_bounds__(256) void slide_update_kernel(float* __restrict__ dst,
                                                           const float* __restrict__ src,
                                                           const float* __restrict__ newv) {
    int idx = blockIdx.x * 256 + threadIdx.x;            // n = NKV*WIN*HD = 2^18
    int h = idx >> 17, p = (idx >> 8) & (WIN - 1), d = idx & (HD - 1);
    float val = (p < WIN - 1) ? src[(h * WIN + p + 1) * HD + d] : newv[h * HD + d];
    dst[idx] = val;
}

// full: dst = src*(1-um[p]) + new[h][d]*um[p]
__global__ __launch_bounds__(256) void full_update_kernel(float* __restrict__ dst,
                                                          const float* __restrict__ src,
                                                          const float* __restrict__ newv,
                                                          const float* __restrict__ um) {
    int idx = blockIdx.x * 256 + threadIdx.x;            // n = NKV*FULL*HD = 2^21
    int h = idx >> 20, p = (idx >> 8) & (FULL - 1), d = idx & (HD - 1);
    float u = um[p];
    dst[idx] = src[idx] * (1.0f - u) + newv[h * HD + d] * u;
}

// ---------------- WMMA flash-decoding attention ----------------
// grid (nsplits, NKV), block 128 (4 waves). Per split of 64 keys:
// scores = Q(4x256,padded 16) @ K^T via v_wmma_f32_16x16x32_f16; online-softmax
// partial; O_partial = P(4xSPLIT) @ V via wmma. Partials merged later.
__global__ __launch_bounds__(128) void attn_split_kernel(const float* __restrict__ qv,
                                                         const float* __restrict__ Kc,
                                                         const float* __restrict__ Vc,
                                                         const float* __restrict__ mask,
                                                         float* __restrict__ m_part,
                                                         float* __restrict__ l_part,
                                                         float* __restrict__ o_part,
                                                         int L, float scale) {
    __shared__ __align__(32) _Float16 Qsh[16 * HD];        // 8 KB
    __shared__ __align__(32) _Float16 Ksh[SPLIT * HD];     // 32 KB
    __shared__ __align__(32) _Float16 Psh[16 * SPLIT];     // 2 KB
    __shared__ __align__(32) _Float16 Vtsh[HD * 32];       // 16 KB (32 keys, transposed)
    __shared__ float Ssh[4 * SPLIT];                       // 1 KB

    const int tid  = threadIdx.x;
    const int wave = tid >> 5, lane = tid & 31;
    const int split = blockIdx.x, g = blockIdx.y;
    const int kb = split * SPLIT;
    const int row = lane & 15;
    const int hi  = (lane < 16) ? 0 : 8;   // unused; A loads also use bh
    const int bh  = (lane < 16) ? 0 : 16;  // B-matrix K-offset per lane half

    // stage Q for this KV group (heads 4g..4g+3), rows 4..15 zero-padded
    for (int idx = tid; idx < 16 * HD; idx += 128) {
        int r = idx >> 8, d = idx & (HD - 1);
        Qsh[idx] = (_Float16)((r < 4) ? qv[(g * 4 + r) * HD + d] : 0.0f);
    }
    // stage 64-key K tile as f16
    for (int idx = tid; idx < SPLIT * HD; idx += 128) {
        int key = idx >> 8, d = idx & (HD - 1);
        Ksh[idx] = (_Float16)Kc[((size_t)g * L + kb + key) * HD + d];
    }
    for (int idx = tid; idx < 16 * SPLIT; idx += 128) Psh[idx] = (_Float16)0.0f;
    __syncthreads();

    // ---- scores: each wave does one 16-key tile; K-dim 256 = 8 chunks of 32
    {
        const int tb = wave * 16;
        v8f acc = {};
        for (int c = 0; c < 8; ++c) {
            int koff = c * 32;
            v16h a, b;
            a = *(const v16h*)&Qsh[row * HD + koff + bh];
            b = *(const v16h*)&Ksh[(tb + row) * HD + koff + bh];
            acc = __builtin_amdgcn_wmma_f32_16x16x32_f16(false, a, false, b,
                                                         (short)0, acc, false, false);
        }
        if (lane < 16)
            for (int r = 0; r < 4; ++r) Ssh[r * SPLIT + tb + lane] = acc[r];
    }
    __syncthreads();

    // ---- partial softmax: wave r handles q-row r over the 64 keys
    {
        int r = wave;
        float s0 = Ssh[r * SPLIT + lane]      * scale + mask[kb + lane];
        float s1 = Ssh[r * SPLIT + lane + 32] * scale + mask[kb + lane + 32];
        float mx = fmaxf(s0, s1);
        for (int off = 16; off > 0; off >>= 1) mx = fmaxf(mx, __shfl_xor(mx, off, 32));
        float p0 = expf(s0 - mx), p1 = expf(s1 - mx);
        float sm = p0 + p1;
        for (int off = 16; off > 0; off >>= 1) sm += __shfl_xor(sm, off, 32);
        Psh[r * SPLIT + lane]      = (_Float16)p0;
        Psh[r * SPLIT + lane + 32] = (_Float16)p1;
        if (lane == 0) {
            int pi = (split * 2 + g) * 4 + r;
            m_part[pi] = mx;
            l_part[pi] = sm;
        }
    }
    __syncthreads();

    // ---- O_partial = P @ V : each wave owns 4 N-tiles of 16, 2 key-chunks of 32
    v8f oacc[4] = {};
    for (int c = 0; c < 2; ++c) {
        int k2 = c * 32;
        for (int idx = tid; idx < 32 * HD; idx += 128) {       // stage V chunk transposed
            int kk = idx >> 8, d = idx & (HD - 1);
            Vtsh[d * 32 + kk] = (_Float16)Vc[((size_t)g * L + kb + k2 + kk) * HD + d];
        }
        __syncthreads();
        v16h a;
        a = *(const v16h*)&Psh[row * SPLIT + k2 + bh];
        for (int t = 0; t < 4; ++t) {
            int col = (wave * 4 + t) * 16 + row;
            v16h b = *(const v16h*)&Vtsh[col * 32 + bh];
            oacc[t] = __builtin_amdgcn_wmma_f32_16x16x32_f16(false, a, false, b,
                                                             (short)0, oacc[t], false, false);
        }
        __syncthreads();
    }
    if (lane < 16) {
        for (int t = 0; t < 4; ++t) {
            int noff = (wave * 4 + t) * 16;
            for (int r = 0; r < 4; ++r)
                o_part[((size_t)(split * 2 + g) * 4 + r) * HD + noff + lane] = oacc[t][r];
        }
    }
}

// merge split partials: 8 blocks (q heads) x 256 threads (HD)
__global__ __launch_bounds__(256) void attn_combine_kernel(float* __restrict__ out,
                                                           const float* __restrict__ m_part,
                                                           const float* __restrict__ l_part,
                                                           const float* __restrict__ o_part,
                                                           int nsplits) {
    int h = blockIdx.x, g = h >> 2, r = h & 3, d = threadIdx.x;
    float mmax = -3.0e38f;
    for (int s = 0; s < nsplits; ++s) mmax = fmaxf(mmax, m_part[(s * 2 + g) * 4 + r]);
    float lsum = 0.f, acc = 0.f;
    for (int s = 0; s < nsplits; ++s) {
        int pi = (s * 2 + g) * 4 + r;
        float w = expf(m_part[pi] - mmax);
        lsum += l_part[pi] * w;
        acc  += o_part[(size_t)pi * HD + d] * w;
    }
    out[h * HD + d] = acc / lsum;
}

// ---------------- host orchestration ----------------
extern "C" void kernel_launch(void* const* d_in, const int* in_sizes, int n_in,
                              void* d_out, int out_size, void* d_ws, size_t ws_size,
                              hipStream_t stream) {
    (void)in_sizes; (void)n_in; (void)out_size; (void)ws_size;
    const float* hidden     = (const float*)d_in[0];
    const float* mask_full  = (const float*)d_in[1];
    const float* mask_slide = (const float*)d_in[2];
    const float* upd_mask   = (const float*)d_in[3];
    const float* pl_raw     = (const float*)d_in[4];
    const float* cos_s      = (const float*)d_in[5];
    const float* sin_s      = (const float*)d_in[6];
    const float* cos_f      = (const float*)d_in[7];
    const float* sin_f      = (const float*)d_in[8];
    const float* Ks_in      = (const float*)d_in[9];
    const float* Vs_in      = (const float*)d_in[10];
    const float* Kf_in      = (const float*)d_in[11];
    const float* Vf_in      = (const float*)d_in[12];
    const float* Wq         = (const float*)d_in[13];
    const float* Wk         = (const float*)d_in[14];
    const float* Wv         = (const float*)d_in[15];
    const float* Wo         = (const float*)d_in[16];
    const float* q_norm_w   = (const float*)d_in[17];
    const float* k_norm_w   = (const float*)d_in[18];
    const float* ln_in_w    = (const float*)d_in[19];
    const float* ln_pa_w    = (const float*)d_in[20];
    const float* ln_pf_w    = (const float*)d_in[21];
    const float* ln_ff_w    = (const float*)d_in[22];
    const float* Wgate      = (const float*)d_in[23];
    const float* Wup        = (const float*)d_in[24];
    const float* Wdown      = (const float*)d_in[25];
    const float* Wplg       = (const float*)d_in[26];
    const float* Wplp       = (const float*)d_in[27];
    const float* ln_pl_w    = (const float*)d_in[28];
    const float* lscalar    = (const float*)d_in[29];
    const float* W_ple      = (const float*)d_in[30];
    const float* ple_nw     = (const float*)d_in[31];

    float* out = (float*)d_out;
    float* ws  = (float*)d_ws;

    // workspace layout (floats); total 174080 floats = 680 KB
    float* hs      = ws;            // 2048
    float* hbuf    = ws + 2048;     // 2048
    float* qbuf    = ws + 4096;     // 2048
    float* kbuf    = ws + 6144;     // 512
    float* vbuf    = ws + 6656;     // 512
    float* attnbuf = ws + 7168;     // 2048
    float* t2048   = ws + 9216;     // 2048
    float* gatebuf = ws + 11264;    // 4096
    float* upbuf   = ws + 15360;    // 4096
    float* actbuf  = ws + 19456;    // 4096
    float* plcbuf  = ws + 23552;    // 8960
    float* pleproj = ws + 32512;    // 8960
    float* plgate  = ws + 41472;    // 256
    float* plgated = ws + 41728;    // 256
    float* m_part  = ws + 41984;    // 512
    float* l_part  = ws + 42496;    // 512
    float* o_part  = ws + 43008;    // 131072

    // output layout: hs | Ks(7,2,512,256) | Vs | Kf(1,2,4096,256) | Vf
    const size_t SKV = (size_t)NKV * WIN * HD;   // 262144
    const size_t FKV = (size_t)NKV * FULL * HD;  // 2097152
    float* outKs = out + H;
    float* outVs = outKs + 7 * SKV;
    float* outKf = outVs + 7 * SKV;
    float* outVf = outKf + 1 * FKV;

    const float PLE_PROJ_SCALE = 0.022097086912079608f;  // 2048^-0.5
    const float ATTN_SCALE = 0.0625f;                    // 256^-0.5

    // hs = hidden; per-layer embedding contributions
    copy_kernel<<<8, 256, 0, stream>>>(hs, hidden, H);
    gemv_kernel<<<(NLT * PLD) / 8, 256, 0, stream>>>(pleproj, hidden, W_ple, H, PLE_PROJ_SCALE);
    ple_kernel<<<NLT, 256, 0, stream>>>(plcbuf, pleproj, ple_nw, pl_raw);

    int si = 0, fi = 0;
    for (int i = 0; i < NL; ++i) {
        const bool is_full = (i % 5 == 4);
        const float* cosv = is_full ? cos_f : cos_s;
        const float* sinv = is_full ? sin_f : sin_s;

        // ---- attention block
        rms_kernel<<<1, 256, 0, stream>>>(hbuf, hs, ln_in_w + i * H, H);
        gemv_kernel<<<(NH * HD) / 8, 256, 0, stream>>>(qbuf, hbuf, Wq + (size_t)i * NH * HD * H, H, 1.f);
        gemv_kernel<<<(NKV * HD) / 8, 256, 0, stream>>>(kbuf, hbuf, Wk + (size_t)i * NKV * HD * H, H, 1.f);
        gemv_kernel<<<(NKV * HD) / 8, 256, 0, stream>>>(vbuf, hbuf, Wv + (size_t)i * NKV * HD * H, H, 1.f);
        qkvpost_kernel<<<12, 256, 0, stream>>>(qbuf, kbuf, vbuf,
                                               q_norm_w + i * HD, k_norm_w + i * HD, cosv, sinv);

        const float* Kc; const float* Vc; const float* mask; int L, nsplits;
        if (is_full) {
            float* dk = outKf + (size_t)fi * FKV;
            float* dv = outVf + (size_t)fi * FKV;
            full_update_kernel<<<FKV / 256, 256, 0, stream>>>(dk, Kf_in + (size_t)fi * FKV, kbuf, upd_mask);
            full_update_kernel<<<FKV / 256, 256, 0, stream>>>(dv, Vf_in + (size_t)fi * FKV, vbuf, upd_mask);
            Kc = dk; Vc = dv; mask = mask_full; L = FULL; nsplits = FULL / SPLIT; ++fi;
        } else {
            float* dk = outKs + (size_t)si * SKV;
            float* dv = outVs + (size_t)si * SKV;
            slide_update_kernel<<<SKV / 256, 256, 0, stream>>>(dk, Ks_in + (size_t)si * SKV, kbuf);
            slide_update_kernel<<<SKV / 256, 256, 0, stream>>>(dv, Vs_in + (size_t)si * SKV, vbuf);
            Kc = dk; Vc = dv; mask = mask_slide; L = WIN; nsplits = WIN / SPLIT; ++si;
        }
        attn_split_kernel<<<dim3(nsplits, NKV), 128, 0, stream>>>(qbuf, Kc, Vc, mask,
                                                                  m_part, l_part, o_part,
                                                                  L, ATTN_SCALE);
        attn_combine_kernel<<<NH, 256, 0, stream>>>(attnbuf, m_part, l_part, o_part, nsplits);
        gemv_kernel<<<H / 8, 256, 0, stream>>>(t2048, attnbuf, Wo + (size_t)i * H * NH * HD, NH * HD, 1.f);
        addrms_kernel<<<1, 256, 0, stream>>>(hs, t2048, ln_pa_w + i * H, H, nullptr, 0);

        // ---- MLP block
        rms_kernel<<<1, 256, 0, stream>>>(hbuf, hs, ln_pf_w + i * H, H);
        gemv_kernel<<<DFF / 8, 256, 0, stream>>>(gatebuf, hbuf, Wgate + (size_t)i * DFF * H, H, 1.f);
        gemv_kernel<<<DFF / 8, 256, 0, stream>>>(upbuf, hbuf, Wup + (size_t)i * DFF * H, H, 1.f);
        gelu_mul_kernel<<<DFF / 256, 256, 0, stream>>>(actbuf, gatebuf, upbuf, DFF);
        gemv_kernel<<<H / 8, 256, 0, stream>>>(t2048, actbuf, Wdown + (size_t)i * H * DFF, DFF, 1.f);
        addrms_kernel<<<1, 256, 0, stream>>>(hs, t2048, ln_ff_w + i * H, H, nullptr, 0);

        // ---- per-layer embedding block (uses raw hs, no pre-norm)
        gemv_kernel<<<PLD / 8, 256, 0, stream>>>(plgate, hs, Wplg + (size_t)i * PLD * H, H, 1.f);
        gelu_mul_kernel<<<1, 256, 0, stream>>>(plgated, plgate, plcbuf + i * PLD, PLD);
        gemv_kernel<<<H / 8, 256, 0, stream>>>(t2048, plgated, Wplp + (size_t)i * H * PLD, PLD, 1.f);
        addrms_kernel<<<1, 256, 0, stream>>>(hs, t2048, ln_pl_w + i * H, H, lscalar, i);
    }
    copy_kernel<<<8, 256, 0, stream>>>(out, hs, H);
}
